// GraphSAGEWithSkip_19284403159492
// MI455X (gfx1250) — compile-verified, matrix operand order checked by reading, not run on hardware
//
#include <hip/hip_runtime.h>
#include <hip/hip_bf16.h>

// ---------------------------------------------------------------------------
// GraphSAGE (3x SAGEConv + BN + skip) inference for MI455X / gfx1250.
// Dense GEMMs: v_wmma_f32_16x16x32_f16 (f32 accumulate), all operands staged
// once as f16 (weights transposed) so fragments are contiguous b128 loads.
// Sparse mean-aggregation: f32 global atomics + one mean->f16 pass.
// ---------------------------------------------------------------------------

namespace {

constexpr int F_IN  = 128;
constexpr int H1    = 256;
constexpr int H2    = 128;
constexpr int D_EMB = 64;
constexpr float BN_EPS = 1e-5f;

typedef __attribute__((ext_vector_type(16))) _Float16 v16h;
typedef __attribute__((ext_vector_type(8)))  _Float16 v8h;
typedef __attribute__((ext_vector_type(8)))  float    v8f;

// ---- WMMA fragment loaders (layouts per CDNA5 ISA §7.12.2) ------
// A (16x32 f16, row-major source): lane = 16*half + m
//   elems 0..7  -> k = k0 + half*8 + j      (contiguous 16B)
//   elems 8..15 -> k = k0 + 16 + half*8 + j (contiguous 16B)
__device__ inline v16h load_frag_a16(const _Float16* __restrict__ A, int lda,
                                     int row0, int k0) {
  const int lane = threadIdx.x & 31;
  const int half = lane >> 4;
  const _Float16* p = A + (size_t)(row0 + (lane & 15)) * lda + k0 + half * 8;
  v8h lo = *(const v8h*)p;
  v8h hi = *(const v8h*)(p + 16);
  return __builtin_shufflevector(lo, hi, 0, 1, 2, 3, 4, 5, 6, 7,
                                 8, 9, 10, 11, 12, 13, 14, 15);
}

// B (32x16 f16) from transposed weights Wt[N x K] (n-major): lane = 16*half+n,
// same contiguous k pattern as A.
__device__ inline v16h load_frag_b16(const _Float16* __restrict__ Wt, int K,
                                     int k0, int n0) {
  const int lane = threadIdx.x & 31;
  const int half = lane >> 4;
  const _Float16* p = Wt + (size_t)(n0 + (lane & 15)) * K + k0 + half * 8;
  v8h lo = *(const v8h*)p;
  v8h hi = *(const v8h*)(p + 16);
  return __builtin_shufflevector(lo, hi, 0, 1, 2, 3, 4, 5, 6, 7,
                                 8, 9, 10, 11, 12, 13, 14, 15);
}

__device__ inline v8f wmma(v16h a, v16h b, v8f c) {
  return __builtin_amdgcn_wmma_f32_16x16x32_f16(false, a, false, b,
                                                (short)0, c, false, false);
}

} // namespace

// ---- utility kernels -------------------------------------------------------

__global__ void fill_zero_kernel(float* __restrict__ p, size_t n) {
  size_t i = (size_t)blockIdx.x * blockDim.x + threadIdx.x;
  if (i < n) p[i] = 0.0f;
}

__global__ void f32_to_f16_kernel(const float* __restrict__ src,
                                  _Float16* __restrict__ dst, long long n) {
  long long i = (long long)blockIdx.x * blockDim.x + threadIdx.x;
  if (i < n) dst[i] = (_Float16)src[i];
}

// W [K x N] f32 row-major  ->  Wt [N x K] f16
__global__ void wt_convert_kernel(const float* __restrict__ W,
                                  _Float16* __restrict__ Wt, int K, int N) {
  int idx = blockIdx.x * blockDim.x + threadIdx.x;
  if (idx >= K * N) return;
  int k = idx / N, n = idx % N;
  Wt[(size_t)n * K + k] = (_Float16)W[idx];
}

// mh[i,f] = f16( sum[i,f] / max(cnt[i],1) )
__global__ void mean_to_f16_kernel(const float* __restrict__ sum,
                                   const float* __restrict__ cnt,
                                   _Float16* __restrict__ mh,
                                   long long total, int dlog2) {
  long long idx = (long long)blockIdx.x * blockDim.x + threadIdx.x;
  if (idx >= total) return;
  int i = (int)(idx >> dlog2);
  mh[idx] = (_Float16)(sum[idx] / fmaxf(cnt[i], 1.0f));
}

// ---- sparse kernels --------------------------------------------------------

__global__ void degree_kernel(const int* __restrict__ dst,
                              float* __restrict__ cnt, int E) {
  int e = blockIdx.x * blockDim.x + threadIdx.x;
  if (e < E) atomicAdd(&cnt[dst[e]], 1.0f);
}

__global__ void scatter_add_f32_kernel(const float* __restrict__ h,
                                       const int* __restrict__ src,
                                       const int* __restrict__ dst,
                                       float* __restrict__ sum,
                                       long long total, int dlog2) {
  long long idx = (long long)blockIdx.x * blockDim.x + threadIdx.x;
  if (idx >= total) return;
  int e = (int)(idx >> dlog2);
  int f = (int)(idx & ((1 << dlog2) - 1));
  atomicAdd(&sum[((size_t)dst[e] << dlog2) + f],
            h[((size_t)src[e] << dlog2) + f]);
}

__global__ void scatter_add_f16_kernel(const _Float16* __restrict__ h,
                                       const int* __restrict__ src,
                                       const int* __restrict__ dst,
                                       float* __restrict__ sum,
                                       long long total, int dlog2) {
  long long idx = (long long)blockIdx.x * blockDim.x + threadIdx.x;
  if (idx >= total) return;
  int e = (int)(idx >> dlog2);
  int f = (int)(idx & ((1 << dlog2) - 1));
  atomicAdd(&sum[((size_t)dst[e] << dlog2) + f],
            (float)h[((size_t)src[e] << dlog2) + f]);
}

// ---- layer 1: h1 = relu( bn1(mean1@W1l + b1l + x@W1r) + x@Wp + bp ) --------
__global__ __launch_bounds__(256) void gemm1_kernel(
    const _Float16* __restrict__ x16, const _Float16* __restrict__ mh,
    const _Float16* __restrict__ WpT, const float* __restrict__ bp,
    const _Float16* __restrict__ W1lT, const float* __restrict__ b1l,
    const _Float16* __restrict__ W1rT,
    const float* __restrict__ g,  const float* __restrict__ bb,
    const float* __restrict__ mm, const float* __restrict__ vv,
    _Float16* __restrict__ h1, int M) {
  const int NT_N = H1 / 16;
  const int wave = blockIdx.x * (blockDim.x >> 5) + (threadIdx.x >> 5);
  if (wave >= (M / 16) * NT_N) return;            // wave-uniform: EXEC full
  const int row0 = (wave / NT_N) * 16;
  const int n0   = (wave % NT_N) * 16;
  const int lane = threadIdx.x & 31;

  v8f accC = {};   // conv branch (gets BN affine)
  v8f accI = {};   // identity branch (input_proj)
#pragma unroll
  for (int k0 = 0; k0 < F_IN; k0 += 32) {
    v16h aM = load_frag_a16(mh,  F_IN, row0, k0);
    v16h aX = load_frag_a16(x16, F_IN, row0, k0);
    accC = wmma(aM, load_frag_b16(W1lT, F_IN, k0, n0), accC);
    accC = wmma(aX, load_frag_b16(W1rT, F_IN, k0, n0), accC);
    accI = wmma(aX, load_frag_b16(WpT,  F_IN, k0, n0), accI);
  }
  const int n = n0 + (lane & 15);
  const float s  = g[n] * rsqrtf(vv[n] + BN_EPS);
  const float t  = bb[n] - mm[n] * s;
  const float bC = b1l[n];
  const float bI = bp[n];
  const int half = lane >> 4;
#pragma unroll
  for (int vi = 0; vi < 8; ++vi) {
    int m = row0 + half * 8 + vi;
    float val = (accC[vi] + bC) * s + t + accI[vi] + bI;
    h1[(size_t)m * H1 + n] = (_Float16)fmaxf(val, 0.0f);
  }
}

// ---- layer 2: h2 = relu( bn2(mean2@W2l + b2l + h1@W2r) + h1@Wsk + bsk ) ----
__global__ __launch_bounds__(256) void gemm2_kernel(
    const _Float16* __restrict__ h1, const _Float16* __restrict__ mh,
    const _Float16* __restrict__ WskT, const float* __restrict__ bsk,
    const _Float16* __restrict__ W2lT, const float* __restrict__ b2l,
    const _Float16* __restrict__ W2rT,
    const float* __restrict__ g,  const float* __restrict__ bb,
    const float* __restrict__ mm, const float* __restrict__ vv,
    _Float16* __restrict__ h2, int M) {
  const int NT_N = H2 / 16;
  const int wave = blockIdx.x * (blockDim.x >> 5) + (threadIdx.x >> 5);
  if (wave >= (M / 16) * NT_N) return;
  const int row0 = (wave / NT_N) * 16;
  const int n0   = (wave % NT_N) * 16;
  const int lane = threadIdx.x & 31;

  v8f accC = {};
  v8f accS = {};
#pragma unroll
  for (int k0 = 0; k0 < H1; k0 += 32) {
    v16h aM = load_frag_a16(mh, H1, row0, k0);
    v16h aH = load_frag_a16(h1, H1, row0, k0);
    accC = wmma(aM, load_frag_b16(W2lT, H1, k0, n0), accC);
    accC = wmma(aH, load_frag_b16(W2rT, H1, k0, n0), accC);
    accS = wmma(aH, load_frag_b16(WskT, H1, k0, n0), accS);
  }
  const int n = n0 + (lane & 15);
  const float s  = g[n] * rsqrtf(vv[n] + BN_EPS);
  const float t  = bb[n] - mm[n] * s;
  const float bC = b2l[n];
  const float bS = bsk[n];
  const int half = lane >> 4;
#pragma unroll
  for (int vi = 0; vi < 8; ++vi) {
    int m = row0 + half * 8 + vi;
    float val = (accC[vi] + bC) * s + t + accS[vi] + bS;
    h2[(size_t)m * H2 + n] = (_Float16)fmaxf(val, 0.0f);
  }
}

// ---- layer 3: emb = mean3@W3l + b3l + h2@W3r (f32 out) ---------------------
__global__ __launch_bounds__(256) void gemm3_kernel(
    const _Float16* __restrict__ h2, const _Float16* __restrict__ mh,
    const _Float16* __restrict__ W3lT, const float* __restrict__ b3l,
    const _Float16* __restrict__ W3rT,
    float* __restrict__ emb, int M) {
  const int NT_N = D_EMB / 16;
  const int wave = blockIdx.x * (blockDim.x >> 5) + (threadIdx.x >> 5);
  if (wave >= (M / 16) * NT_N) return;
  const int row0 = (wave / NT_N) * 16;
  const int n0   = (wave % NT_N) * 16;
  const int lane = threadIdx.x & 31;

  v8f acc = {};
#pragma unroll
  for (int k0 = 0; k0 < H2; k0 += 32) {
    v16h aM = load_frag_a16(mh, H2, row0, k0);
    v16h aH = load_frag_a16(h2, H2, row0, k0);
    acc = wmma(aM, load_frag_b16(W3lT, H2, k0, n0), acc);
    acc = wmma(aH, load_frag_b16(W3rT, H2, k0, n0), acc);
  }
  const int n = n0 + (lane & 15);
  const float bC = b3l[n];
  const int half = lane >> 4;
#pragma unroll
  for (int vi = 0; vi < 8; ++vi) {
    int m = row0 + half * 8 + vi;
    emb[(size_t)m * D_EMB + n] = acc[vi] + bC;
  }
}

// ---- classifier: logits = emb @ Wc + bc  (64 -> 2) -------------------------
__global__ void classifier_kernel(const float* __restrict__ emb,
                                  const float* __restrict__ Wc,
                                  const float* __restrict__ bc,
                                  float* __restrict__ logits, int M) {
  int i = blockIdx.x * blockDim.x + threadIdx.x;
  if (i >= M) return;
  const float* e = emb + (size_t)i * D_EMB;
  float a0 = bc[0], a1 = bc[1];
#pragma unroll
  for (int k = 0; k < D_EMB; ++k) {
    float v = e[k];
    a0 += v * Wc[k * 2 + 0];
    a1 += v * Wc[k * 2 + 1];
  }
  logits[(size_t)i * 2 + 0] = a0;
  logits[(size_t)i * 2 + 1] = a1;
}

// ---------------------------------------------------------------------------

extern "C" void kernel_launch(void* const* d_in, const int* in_sizes, int n_in,
                              void* d_out, int out_size, void* d_ws, size_t ws_size,
                              hipStream_t stream) {
  const float* x    = (const float*)d_in[0];
  const int*   ei   = (const int*)  d_in[1];
  const float* Wp   = (const float*)d_in[2];
  const float* bp   = (const float*)d_in[3];
  const float* W1l  = (const float*)d_in[4];
  const float* b1l  = (const float*)d_in[5];
  const float* W1r  = (const float*)d_in[6];
  const float* bn1g = (const float*)d_in[7];
  const float* bn1b = (const float*)d_in[8];
  const float* bn1m = (const float*)d_in[9];
  const float* bn1v = (const float*)d_in[10];
  const float* Wsk  = (const float*)d_in[11];
  const float* bsk  = (const float*)d_in[12];
  const float* W2l  = (const float*)d_in[13];
  const float* b2l  = (const float*)d_in[14];
  const float* W2r  = (const float*)d_in[15];
  const float* bn2g = (const float*)d_in[16];
  const float* bn2b = (const float*)d_in[17];
  const float* bn2m = (const float*)d_in[18];
  const float* bn2v = (const float*)d_in[19];
  const float* W3l  = (const float*)d_in[20];
  const float* b3l  = (const float*)d_in[21];
  const float* W3r  = (const float*)d_in[22];
  const float* Wc   = (const float*)d_in[23];
  const float* bc   = (const float*)d_in[24];

  const int M = in_sizes[0] / F_IN;     // 50000 nodes (multiple of 16)
  const int E = in_sizes[1] / 2;        // 800000 edges
  const int* src = ei;
  const int* dst = ei + E;

  // Workspace layout:
  //   f32: cnt[M] | sum[256*M]
  //   f16: x16[128*M] | mh[256*M] | h1[256*M] | h2[128*M] | Wt buffers
  float*    cnt = (float*)d_ws;
  float*    sum = cnt + (size_t)M;
  _Float16* x16 = (_Float16*)(sum + (size_t)256 * M);
  _Float16* mh  = x16 + (size_t)128 * M;
  _Float16* h1h = mh  + (size_t)256 * M;
  _Float16* h2h = h1h + (size_t)256 * M;
  _Float16* WpT  = h2h + (size_t)128 * M;   // 256x128
  _Float16* W1lT = WpT  + 32768;            // 256x128
  _Float16* W1rT = W1lT + 32768;            // 256x128
  _Float16* WskT = W1rT + 32768;            // 128x256
  _Float16* W2lT = WskT + 32768;            // 128x256
  _Float16* W2rT = W2lT + 32768;            // 128x256
  _Float16* W3lT = W2rT + 32768;            // 64x128
  _Float16* W3rT = W3lT + 8192;             // 64x128

  float* logits = (float*)d_out;                   // M x 2
  float* emb    = (float*)d_out + (size_t)M * 2;   // M x 64

  auto zero = [&](float* p, size_t n) {
    fill_zero_kernel<<<(unsigned)((n + 255) / 256), 256, 0, stream>>>(p, n);
  };
  auto wconv = [&](const float* W, _Float16* Wt, int K, int N) {
    wt_convert_kernel<<<(K * N + 255) / 256, 256, 0, stream>>>(W, Wt, K, N);
  };

  // --- one-time operand staging (f16) --------------------------------------
  {
    long long n = (long long)M * F_IN;
    f32_to_f16_kernel<<<(unsigned)((n + 255) / 256), 256, 0, stream>>>(x, x16, n);
  }
  wconv(Wp,  WpT,  F_IN, H1);
  wconv(W1l, W1lT, F_IN, H1);
  wconv(W1r, W1rT, F_IN, H1);
  wconv(Wsk, WskT, H1, H2);
  wconv(W2l, W2lT, H1, H2);
  wconv(W2r, W2rT, H1, H2);
  wconv(W3l, W3lT, H2, D_EMB);
  wconv(W3r, W3rT, H2, D_EMB);

  // --- phase 1: degrees + conv1 aggregation + fused GEMM1 -------------------
  zero(cnt, (size_t)M * (1 + 256));
  degree_kernel<<<(E + 255) / 256, 256, 0, stream>>>(dst, cnt, E);
  {
    long long total = (long long)E << 7;           // E * 128
    scatter_add_f32_kernel<<<(unsigned)((total + 255) / 256), 256, 0, stream>>>(
        x, src, dst, sum, total, 7);
    long long mt = (long long)M << 7;
    mean_to_f16_kernel<<<(unsigned)((mt + 255) / 256), 256, 0, stream>>>(
        sum, cnt, mh, mt, 7);
  }
  {
    int waves = (M / 16) * (H1 / 16);
    gemm1_kernel<<<(waves + 7) / 8, 256, 0, stream>>>(
        x16, mh, WpT, bp, W1lT, b1l, W1rT, bn1g, bn1b, bn1m, bn1v, h1h, M);
  }

  // --- phase 2: conv2 aggregation + fused GEMM2 -----------------------------
  zero(sum, (size_t)M * 256);
  {
    long long total = (long long)E << 8;           // E * 256
    scatter_add_f16_kernel<<<(unsigned)((total + 255) / 256), 256, 0, stream>>>(
        h1h, src, dst, sum, total, 8);
    long long mt = (long long)M << 8;
    mean_to_f16_kernel<<<(unsigned)((mt + 255) / 256), 256, 0, stream>>>(
        sum, cnt, mh, mt, 8);
  }
  {
    int waves = (M / 16) * (H2 / 16);
    gemm2_kernel<<<(waves + 7) / 8, 256, 0, stream>>>(
        h1h, mh, WskT, bsk, W2lT, b2l, W2rT, bn2g, bn2b, bn2m, bn2v, h2h, M);
  }

  // --- phase 3: conv3 aggregation + GEMM3 + classifier ----------------------
  zero(sum, (size_t)M * 128);
  {
    long long total = (long long)E << 7;           // E * 128
    scatter_add_f16_kernel<<<(unsigned)((total + 255) / 256), 256, 0, stream>>>(
        h2h, src, dst, sum, total, 7);
    long long mt = (long long)M << 7;
    mean_to_f16_kernel<<<(unsigned)((mt + 255) / 256), 256, 0, stream>>>(
        sum, cnt, mh, mt, 7);
  }
  {
    int waves = (M / 16) * (D_EMB / 16);
    gemm3_kernel<<<(waves + 7) / 8, 256, 0, stream>>>(
        h2h, mh, W3lT, b3l, W3rT, emb, M);
  }
  classifier_kernel<<<(M + 255) / 256, 256, 0, stream>>>(emb, Wc, bc, logits, M);
}
